// Model_84026740179765
// MI455X (gfx1250) — matrix operand order repaired and checked by reading, not compiled, not run on hardware
//
#include <hip/hip_runtime.h>

// ============================================================================
// MI455X (gfx1250) neighborhood attention; all GEMM work on
// v_wmma_f32_16x16x32_bf16 (bf16 in / f32 accumulate).
//   - x and weights converted to bf16 ONCE; all GEMMs run the bf16-A path
//   - DPP16 row_ror butterfly for softmax row-max (no LDS, no waits)
//   - softmax row-SUM on the matrix unit (5th all-ones V tile = flash "l")
//   - mask bits packed as [H, chunk, L]: 2 uint4 loads per 32-key chunk
//   - BLOCK-SPARSE chunk skipping: all-masked 32-key chunks are skipped at
//     wave granularity (__any) and staging+barriers at block granularity
//     (__syncthreads_or) -- exact, since skipped chunks leave state unchanged
//
// Inputs: x f32[128,512,512], Wq/Wk/Wv/Wo f32[512,512], na_mask bool[8,512,512]
// Output: f32 [128,512,512]
// Workspace: [0,2MB) W* bf16 | [2,66) x bf16 | [66,130) Q | [130,194) K
//            [194,258) V | [258,322) attn-out bf16 | [322MB,+256KB) mask bits
// ============================================================================

typedef __attribute__((ext_vector_type(16))) __bf16 v16bf;
typedef __attribute__((ext_vector_type(8)))  float  v8f;

#define D_MODEL 512
#define SEQ_L   512
#define N_HEADS 8
#define HEAD_E  64

__device__ __forceinline__ unsigned pack_bf16x2(float lo, float hi) {
  union { __bf16 h[2]; unsigned u; } x;
  x.h[0] = (__bf16)lo; x.h[1] = (__bf16)hi;    // native v_cvt (RTNE)
  return x.u;
}

// 16-lane (DPP row) butterfly max: all lanes of each 16-lane row get the max.
#define DPP_STEP_MAX(v, ctrl)                                                   \
  v = fmaxf(v, __int_as_float(__builtin_amdgcn_mov_dpp(__float_as_int(v),       \
                                                       ctrl, 0xF, 0xF, true)))
__device__ __forceinline__ float red16_max(float v) {
  DPP_STEP_MAX(v, 0x128);  // row_ror:8
  DPP_STEP_MAX(v, 0x124);  // row_ror:4
  DPP_STEP_MAX(v, 0x122);  // row_ror:2
  DPP_STEP_MAX(v, 0x121);  // row_ror:1
  return v;
}

union Frag { v16bf v; uint4 q[2]; };

#define WMMA_BF16(a, b, c)                                                      \
  __builtin_amdgcn_wmma_f32_16x16x32_bf16(false, (a), false, (b), (short)0,     \
                                          (c), false, false)
#define SCHED_FENCE() asm volatile("" ::: "memory")

// ---------------------------------------------------------------------------
// vectorized f32 -> bf16 (4 elements / thread)
__global__ __launch_bounds__(256) void cvt_f32_bf16_x4(const float4* __restrict__ in,
                                                       uint2* __restrict__ out, int n4) {
  int i = blockIdx.x * blockDim.x + threadIdx.x;
  if (i >= n4) return;
  const float4 f = in[i];
  uint2 w;
  w.x = pack_bf16x2(f.x, f.y);
  w.y = pack_bf16x2(f.z, f.w);
  out[i] = w;
}

// Pack bool mask [H,L,L] into u32 words laid out [H, chunk, L] so the 8
// row-words one lane needs are contiguous (two uint4 loads per chunk).
__global__ __launch_bounds__(256) void pack_mask(const unsigned char* __restrict__ m,
                                                 unsigned* __restrict__ out, int nwords) {
  int i = blockIdx.x * blockDim.x + threadIdx.x;   // i = (h*512 + row)*16 + chunk
  if (i >= nwords) return;
  const unsigned* s4 = (const unsigned*)(m + (size_t)i * 32);
  unsigned w = 0;
#pragma unroll
  for (int q = 0; q < 8; ++q) {
    const unsigned v = s4[q];
    w |= ((v & 0x000000FFu) ? 1u : 0u) << (4 * q + 0);
    w |= ((v & 0x0000FF00u) ? 1u : 0u) << (4 * q + 1);
    w |= ((v & 0x00FF0000u) ? 1u : 0u) << (4 * q + 2);
    w |= ((v & 0xFF000000u) ? 1u : 0u) << (4 * q + 3);
  }
  const int chunk = i & 15, row = (i >> 4) & 511, h = i >> 13;
  out[((size_t)h * 16 + chunk) * SEQ_L + row] = w;
}

// ---------------------------------------------------------------------------
// C[M,512] = A[M,512] @ W^T, A and W bf16. Block tile 128x64, BK=32, 256 thr
// = 8 waves (4x2), each wave 2x2 WMMA tiles. Double-buffered LDS; next
// K-slice held in VGPRs while WMMAs run.
// ---------------------------------------------------------------------------
template <bool STORE_BF16>
__global__ __launch_bounds__(256)
void gemm_bf16(const unsigned short* __restrict__ A, const unsigned short* __restrict__ W,
               void* __restrict__ Cv, float scale) {
  __shared__ unsigned short As[2][128][40];   // 80B rows: 16B aligned, bank-spread
  __shared__ unsigned short Bs[2][64][40];

  const int tid  = threadIdx.x;
  const int lane = tid & 31, wave = tid >> 5;
  const int wm = wave & 3, wn = wave >> 2;
  const int ln = lane & 15, lh = lane >> 4;
  const int m0 = blockIdx.x * 128;
  const int n0 = blockIdx.y * 64;

  v8f acc[2][2] = {};
  uint4 ua0, ua1, fb;

  auto gload = [&](int k0) {
    const uint4* src = (const uint4*)(A + (size_t)(m0 + (tid >> 1)) * D_MODEL + k0);
    ua0 = src[(tid & 1) * 2]; ua1 = src[(tid & 1) * 2 + 1];
    fb  = ((const uint4*)(W + (size_t)(n0 + (tid >> 2)) * D_MODEL + k0))[tid & 3];
  };
  auto sstore = [&](int buf) {
    const int row = tid >> 1, c = (tid & 1) * 2;
    *(uint4*)&As[buf][row][c * 8]       = ua0;
    *(uint4*)&As[buf][row][(c + 1) * 8] = ua1;
    *(uint4*)&Bs[buf][tid >> 2][(tid & 3) * 8] = fb;
  };
  auto compute = [&](int buf) {
    Frag a[2], b[2];
#pragma unroll
    for (int mi = 0; mi < 2; ++mi) {
      const unsigned short* rp = &As[buf][wm * 32 + mi * 16 + ln][0];
      a[mi].q[0] = *(const uint4*)(rp + lh * 8);
      a[mi].q[1] = *(const uint4*)(rp + 16 + lh * 8);
    }
#pragma unroll
    for (int ni = 0; ni < 2; ++ni) {
      const unsigned short* rp = &Bs[buf][wn * 32 + ni * 16 + ln][0];
      b[ni].q[0] = *(const uint4*)(rp + lh * 8);
      b[ni].q[1] = *(const uint4*)(rp + 16 + lh * 8);
    }
    SCHED_FENCE();                     // keep the 8 ds_loads ahead of WMMAs
#pragma unroll
    for (int mi = 0; mi < 2; ++mi)
#pragma unroll
      for (int ni = 0; ni < 2; ++ni)
        acc[mi][ni] = WMMA_BF16(a[mi].v, b[ni].v, acc[mi][ni]);
  };

  gload(0);
  sstore(0);
  __syncthreads();

#pragma unroll 1
  for (int kk = 0; kk < D_MODEL / 32; kk += 2) {
    if (kk + 1 < D_MODEL / 32) gload((kk + 1) * 32);
    compute(0);
    if (kk + 1 < D_MODEL / 32) sstore(1);
    __syncthreads();
    if (kk + 2 < D_MODEL / 32) gload((kk + 2) * 32);
    compute(1);
    if (kk + 2 < D_MODEL / 32) sstore(0);
    __syncthreads();
  }

#pragma unroll
  for (int mi = 0; mi < 2; ++mi)
#pragma unroll
    for (int ni = 0; ni < 2; ++ni) {
      const int col = n0 + wn * 32 + ni * 16 + ln;
#pragma unroll
      for (int r = 0; r < 8; ++r) {
        const int row = m0 + wm * 32 + mi * 16 + lh * 8 + r;
        const float v = acc[mi][ni][r] * scale;
        if constexpr (STORE_BF16)
          *(__bf16*)&((unsigned short*)Cv)[(size_t)row * D_MODEL + col] = (__bf16)v;
        else
          ((float*)Cv)[(size_t)row * D_MODEL + col] = v;
      }
    }
}

// ---------------------------------------------------------------------------
// Masked flash attention with block-sparse chunk skipping.
// 128 thr = 4 waves, each wave 16 query rows of one (b,h). K row-major, V
// transposed in LDS; Vt rows 64..79 == 1.0 so tile 4 accumulates rowsum(P).
// ---------------------------------------------------------------------------
__global__ __launch_bounds__(128)
void na_attention(const unsigned short* __restrict__ Q,   // pre-scaled by 1/8
                  const unsigned short* __restrict__ K,
                  const unsigned short* __restrict__ V,
                  const unsigned* __restrict__ maskbits,  // [H,16,L] u32
                  unsigned short* __restrict__ O) {       // [B,L,D] bf16
  __shared__ unsigned short Qs[64][72];
  __shared__ unsigned short Ks[32][72];
  __shared__ unsigned short Vt[80][40];        // Vt[e][s]; rows 64..79 == 1.0
  __shared__ unsigned short Ps[4][16][40];     // wave-private P scratch

  const int tid  = threadIdx.x;
  const int lane = tid & 31, wave = tid >> 5;
  const int ln = lane & 15, lh = lane >> 4;
  const int h = blockIdx.y, b = blockIdx.z;
  const int q0 = blockIdx.x * 64;

#pragma unroll
  for (int p = 0; p < 3; ++p) {                // ones rows (16 x 20 dwords)
    const int i = p * 128 + tid;
    if (i < 320) ((unsigned*)&Vt[64 + i / 20][0])[i % 20] = 0x3F803F80u;
  }
#pragma unroll
  for (int p = 0; p < 4; ++p) {                // stage Q (64x64) once
    const int idx = p * 128 + tid;
    const int row = idx >> 3, c = idx & 7;
    const uint4 d = *(const uint4*)(Q + ((size_t)b * SEQ_L + q0 + row) * D_MODEL +
                                    h * HEAD_E + c * 8);
    *(uint4*)&Qs[row][c * 8] = d;
  }
  __syncthreads();

  Frag aq[2];
  {
    const unsigned short* rp = &Qs[wave * 16 + ln][0];
#pragma unroll
    for (int c = 0; c < 2; ++c) {
      aq[c].q[0] = *(const uint4*)(rp + c * 32 + lh * 8);
      aq[c].q[1] = *(const uint4*)(rp + c * 32 + 16 + lh * 8);
    }
  }

  v8f o[5] = {};                     // o[0..3]=output tiles, o[4]=rowsum "l"
  float mrow[8];
#pragma unroll
  for (int r = 0; r < 8; ++r) mrow[r] = -1.0e30f;   // finite sentinel
  const unsigned* mbase = maskbits + (size_t)h * 16 * SEQ_L + q0 + wave * 16 + lh * 8;

  uint4 kreg0, kreg1, vreg0, vreg1;
  const int krow = tid >> 2, kc = tid & 3;
  const int sp = tid >> 3, vc = tid & 7;
  auto kv_gload = [&](int s0) {
    const uint4* ks = (const uint4*)(K + ((size_t)b * SEQ_L + s0 + krow) * D_MODEL + h * HEAD_E);
    kreg0 = ks[kc * 2]; kreg1 = ks[kc * 2 + 1];
    vreg0 = *((const uint4*)(V + ((size_t)b * SEQ_L + s0 + 2 * sp) * D_MODEL + h * HEAD_E) + vc);
    vreg1 = *((const uint4*)(V + ((size_t)b * SEQ_L + s0 + 2 * sp + 1) * D_MODEL + h * HEAD_E) + vc);
  };
  auto kv_sstore = [&]() {
    *(uint4*)&Ks[krow][kc * 16]     = kreg0;
    *(uint4*)&Ks[krow][kc * 16 + 8] = kreg1;
    const unsigned short* e0 = (const unsigned short*)&vreg0;
    const unsigned short* e1 = (const unsigned short*)&vreg1;
#pragma unroll
    for (int j = 0; j < 8; ++j) {
      const unsigned w = (unsigned)e0[j] | ((unsigned)e1[j] << 16);
      *(unsigned*)&Vt[vc * 8 + j][2 * sp] = w;
    }
  };

  kv_gload(0);

#pragma unroll 1
  for (int s0 = 0; s0 < SEQ_L; s0 += 32) {
    // mask words: 8 contiguous u32 -> two uint4 loads
    const uint4 mw0 = *(const uint4*)(mbase + (size_t)(s0 >> 5) * SEQ_L);
    const uint4 mw1 = *(const uint4*)(mbase + (size_t)(s0 >> 5) * SEQ_L + 4);
    const unsigned wor = mw0.x | mw0.y | mw0.z | mw0.w | mw1.x | mw1.y | mw1.z | mw1.w;
    const int wave_need  = __any((int)(wor != 0));          // wave-uniform
    const int block_need = __syncthreads_or(wave_need);     // block-uniform

    if (block_need) {
      kv_sstore();
      __syncthreads();
    }
    if (s0 + 32 < SEQ_L) kv_gload(s0 + 32);    // keep the pipeline primed

    if (block_need) {
      if (wave_need) {
        const unsigned wr[8] = {mw0.x, mw0.y, mw0.z, mw0.w, mw1.x, mw1.y, mw1.z, mw1.w};

        Frag bk[2][2];
#pragma unroll
        for (int st = 0; st < 2; ++st) {
          const unsigned short* rp = &Ks[st * 16 + ln][0];
#pragma unroll
          for (int c = 0; c < 2; ++c) {
            bk[st][c].q[0] = *(const uint4*)(rp + c * 32 + lh * 8);
            bk[st][c].q[1] = *(const uint4*)(rp + c * 32 + 16 + lh * 8);
          }
        }
        SCHED_FENCE();
        v8f sc0 = {}, sc1 = {};
        sc0 = WMMA_BF16(aq[0].v, bk[0][0].v, sc0);
        sc1 = WMMA_BF16(aq[0].v, bk[1][0].v, sc1);
        sc0 = WMMA_BF16(aq[1].v, bk[0][1].v, sc0);
        sc1 = WMMA_BF16(aq[1].v, bk[1][1].v, sc1);

        // streaming softmax: -inf mask bias, DPP16 max, WMMA row-sum
#pragma unroll
        for (int r = 0; r < 8; ++r) {
          const unsigned w = wr[r];
          const float s0v = ((w >> ln) & 1u)        ? sc0[r] : -__builtin_inff();
          const float s1v = ((w >> (16 + ln)) & 1u) ? sc1[r] : -__builtin_inff();
          float mx = red16_max(fmaxf(s0v, s1v));
          mx = fmaxf(mx, mrow[r]);                  // mx >= -1e30: finite
          const float alpha = __expf(mrow[r] - mx);
          const float p0 = __expf(s0v - mx);        // exp(-inf)=0 masks
          const float p1 = __expf(s1v - mx);
          mrow[r] = mx;
#pragma unroll
          for (int et = 0; et < 5; ++et) o[et][r] *= alpha;   // incl. "l" tile
          ((__bf16*)&Ps[wave][lh * 8 + r][0])[ln]      = (__bf16)p0;
          ((__bf16*)&Ps[wave][lh * 8 + r][0])[16 + ln] = (__bf16)p1;
        }
        asm volatile("s_wait_dscnt 0" ::: "memory");   // wave-internal Ps RAW

        Frag pa, bv[5];
        {
          const unsigned short* rp = &Ps[wave][ln][0];
          pa.q[0] = *(const uint4*)(rp + lh * 8);
          pa.q[1] = *(const uint4*)(rp + 16 + lh * 8);
        }
#pragma unroll
        for (int et = 0; et < 5; ++et) {
          const unsigned short* rp = &Vt[et * 16 + ln][0];
          bv[et].q[0] = *(const uint4*)(rp + lh * 8);
          bv[et].q[1] = *(const uint4*)(rp + 16 + lh * 8);
        }
        SCHED_FENCE();
#pragma unroll
        for (int et = 0; et < 5; ++et)
          o[et] = WMMA_BF16(pa.v, bv[et].v, o[et]);
      }
      __syncthreads();
    }
  }

  // ---- normalize by l (tile 4; all lanes hold the row-sum) + store bf16 ----
#pragma unroll
  for (int r = 0; r < 8; ++r) {
    const float l = o[4][r];
    const float inv = (l > 0.f) ? 1.f / l : 0.f;
    const int grow = q0 + wave * 16 + lh * 8 + r;
#pragma unroll
    for (int et = 0; et < 4; ++et) {
      const int col = h * HEAD_E + et * 16 + ln;
      *(__bf16*)&O[((size_t)b * SEQ_L + grow) * D_MODEL + col] = (__bf16)(o[et][r] * inv);
    }
  }
}

// ---------------------------------------------------------------------------
extern "C" void kernel_launch(void* const* d_in, const int* in_sizes, int n_in,
                              void* d_out, int out_size, void* d_ws, size_t ws_size,
                              hipStream_t stream) {
  const float*         x    = (const float*)d_in[0];
  const float*         Wq   = (const float*)d_in[1];
  const float*         Wk   = (const float*)d_in[2];
  const float*         Wv   = (const float*)d_in[3];
  const float*         Wo   = (const float*)d_in[4];
  const unsigned char* mask = (const unsigned char*)d_in[5];

  char* ws = (char*)d_ws;
  const size_t MB = 1024ull * 1024ull;
  unsigned short* Wq_bf = (unsigned short*)(ws + 0 * MB / 2);
  unsigned short* Wk_bf = (unsigned short*)(ws + 1 * MB / 2);
  unsigned short* Wv_bf = (unsigned short*)(ws + 2 * MB / 2);
  unsigned short* Wo_bf = (unsigned short*)(ws + 3 * MB / 2);
  unsigned short* Xbf   = (unsigned short*)(ws + 2 * MB);
  unsigned short* Qbf   = (unsigned short*)(ws + 66 * MB);
  unsigned short* Kbf   = (unsigned short*)(ws + 130 * MB);
  unsigned short* Vbf   = (unsigned short*)(ws + 194 * MB);
  unsigned short* Abf   = (unsigned short*)(ws + 258 * MB);
  unsigned*       Mbits = (unsigned*)(ws + 322 * MB);

  const int WN4 = D_MODEL * D_MODEL / 4;            // 65536
  cvt_f32_bf16_x4<<<WN4 / 256, 256, 0, stream>>>((const float4*)Wq, (uint2*)Wq_bf, WN4);
  cvt_f32_bf16_x4<<<WN4 / 256, 256, 0, stream>>>((const float4*)Wk, (uint2*)Wk_bf, WN4);
  cvt_f32_bf16_x4<<<WN4 / 256, 256, 0, stream>>>((const float4*)Wv, (uint2*)Wv_bf, WN4);
  cvt_f32_bf16_x4<<<WN4 / 256, 256, 0, stream>>>((const float4*)Wo, (uint2*)Wo_bf, WN4);
  const int XN4 = 128 * SEQ_L * D_MODEL / 4;        // 8388608
  cvt_f32_bf16_x4<<<XN4 / 256, 256, 0, stream>>>((const float4*)x, (uint2*)Xbf, XN4);

  const int NW = N_HEADS * SEQ_L * (SEQ_L / 32);    // 65536 mask words
  pack_mask<<<NW / 256, 256, 0, stream>>>(mask, Mbits, NW);

  const dim3 ggrid(65536 / 128, D_MODEL / 64);      // (512, 8)
  gemm_bf16<true><<<ggrid, 256, 0, stream>>>(Xbf, Wq_bf, Qbf, 0.125f);
  gemm_bf16<true><<<ggrid, 256, 0, stream>>>(Xbf, Wk_bf, Kbf, 1.0f);
  gemm_bf16<true><<<ggrid, 256, 0, stream>>>(Xbf, Wv_bf, Vbf, 1.0f);

  const dim3 agrid(SEQ_L / 64, N_HEADS, 128);
  na_attention<<<agrid, 128, 0, stream>>>(Qbf, Kbf, Vbf, Mbits, Abf);

  gemm_bf16<false><<<ggrid, 256, 0, stream>>>(Abf, Wo_bf, (float*)d_out, 1.0f);
}
